// FlashMultiheadAttention_3315714752642
// MI455X (gfx1250) — matrix operand order repaired and checked by reading, not compiled, not run on hardware
//
#include <hip/hip_runtime.h>
#include <stdint.h>

// Problem sizes (fixed by the reference)
#define B_   2
#define D_   1024
#define S_   2048
#define U_   1024
#define H_   16
#define HD_  64
#define NROW (B_ * S_)   // 4096 rows of the "token" matrix

typedef __attribute__((ext_vector_type(16))) __bf16 bf16x16;
typedef __attribute__((ext_vector_type(8)))  float  f32x8;

union AFrag {           // 32 bytes = 8 VGPRs, holds one WMMA bf16 operand
    bf16x16 v;
    uint4   q[2];
};

static __device__ __forceinline__ uint16_t f2bf(float f) {
    uint32_t x = __float_as_uint(f);
    x += 0x7FFFu + ((x >> 16) & 1u);   // round-to-nearest-even
    return (uint16_t)(x >> 16);
}

static __device__ __forceinline__ f32x8 fzero() {
    f32x8 z = {0.f, 0.f, 0.f, 0.f, 0.f, 0.f, 0.f, 0.f};
    return z;
}

static __device__ __forceinline__ f32x8 wmma_bf16(bf16x16 a, bf16x16 b, f32x8 c) {
    // v_wmma_f32_16x16x32_bf16  (probe-confirmed 8-arg signature)
    return __builtin_amdgcn_wmma_f32_16x16x32_bf16(false, a, false, b, (short)0, c, false, false);
}

// ---------------------------------------------------------------------------
// Kernel 1: (B, D, S) fp32 -> (B, S, D) fp32 (residual) + bf16 (GEMM input)
// ---------------------------------------------------------------------------
__global__ void k_transpose(const float* __restrict__ in,
                            float* __restrict__ xf, uint16_t* __restrict__ xb) {
    __shared__ float tile[32][33];
    int b  = blockIdx.z;
    int d0 = blockIdx.x * 32;
    int s0 = blockIdx.y * 32;
    int tx = threadIdx.x, ty = threadIdx.y;      // blockDim = (32, 8)
    const float* src = in + (size_t)b * D_ * S_;
    #pragma unroll
    for (int i = 0; i < 32; i += 8)
        tile[ty + i][tx] = src[(size_t)(d0 + ty + i) * S_ + (s0 + tx)];
    __syncthreads();
    float*    df = xf + (size_t)b * S_ * D_;
    uint16_t* db = xb + (size_t)b * S_ * D_;
    #pragma unroll
    for (int i = 0; i < 32; i += 8) {
        float v = tile[tx][ty + i];
        size_t o = (size_t)(s0 + ty + i) * D_ + (d0 + tx);
        df[o] = v;
        db[o] = f2bf(v);
    }
}

// ---------------------------------------------------------------------------
// Kernel 2: fp32 -> bf16 weight cast
// ---------------------------------------------------------------------------
__global__ void k_cast(const float* __restrict__ w, uint16_t* __restrict__ o, int n) {
    int i = blockIdx.x * blockDim.x + threadIdx.x;
    if (i < n) o[i] = f2bf(w[i]);
}

// ---------------------------------------------------------------------------
// Kernel 3: bf16 WMMA GEMM, 1024-deep K, with fused epilogues.
//   mode 0: Q/V  -> bf16 (B,H,S,hd), relu, *scale
//   mode 1: K    -> bf16 (B,H,hd,S)  (transposed for attention B-fragments)
//   mode 2: out-proj -> fp32 (NROW, D), +bias +residual
// Workgroup: 256 thr = 8 waves in 2x4; WG tile 64(M) x 256(N); wave = 32x64.
// B tile (32K x 256N) staged into LDS via CDNA5 async LDS-DMA loads
// (GLOBAL_LOAD_ASYNC_TO_LDS_B128, tracked by ASYNCcnt) — no VGPR round-trip.
// ---------------------------------------------------------------------------
__global__ void __launch_bounds__(256)
k_gemm(const uint16_t* __restrict__ X, const uint16_t* __restrict__ W,
       const float* __restrict__ bias, const float* __restrict__ residual,
       uint16_t* __restrict__ out_bf, float* __restrict__ out_f,
       int mode, float scale) {
    __shared__ __align__(16) uint16_t lb[32][256];      // 16 KB

    const int wid  = threadIdx.x >> 5;
    const int lane = threadIdx.x & 31;
    const int wr = wid >> 2;                 // 0..1
    const int wc = wid & 3;                  // 0..3
    const int m_base  = blockIdx.y * 64 + wr * 32;
    const int n_block = blockIdx.x * 256;
    const int n_base  = n_block + wc * 64;

    f32x8 acc[2][4];
    #pragma unroll
    for (int i = 0; i < 2; ++i)
        #pragma unroll
        for (int j = 0; j < 4; ++j) acc[i][j] = fzero();

    const int mrow = m_base + (lane & 15);
    const int kb   = (lane >> 4) * 8;        // A-frag K sub-offset (0 or 8)

    // Per-thread slice of the cooperative B-tile copy: 32 bf16 = 64 B.
    const int kr  = threadIdx.x >> 3;              // 0..31 (K row)
    const int nc0 = (threadIdx.x & 7) * 32;        // column offset in tile
    // Flat->LDS: the LDS address is the low 32 bits of the flat address (ISA 10.2).
    const uint32_t ldsa = (uint32_t)(uintptr_t)&lb[kr][nc0];

    for (int kk = 0; kk < D_; kk += 32) {
        __syncthreads();                           // WAR: prior reads done
        {
            const uint16_t* src = W + (size_t)(kk + kr) * U_ + n_block + nc0;
            // Async LDS-DMA: INST_OFFSET is applied to BOTH the global and the
            // LDS address (ISA 10.7 pseudocode), so 0/16/32/48 copies 64 B.
            asm volatile(
                "global_load_async_to_lds_b128 %0, %1, off\n\t"
                "global_load_async_to_lds_b128 %0, %1, off offset:16\n\t"
                "global_load_async_to_lds_b128 %0, %1, off offset:32\n\t"
                "global_load_async_to_lds_b128 %0, %1, off offset:48"
                :: "v"(ldsa), "v"(src) : "memory");
            asm volatile("s_wait_asynccnt 0" ::: "memory");
        }
        __syncthreads();                           // RAW: whole tile visible

        AFrag a0, a1;
        const uint16_t* x0 = X + (size_t)mrow * D_ + kk + kb;
        a0.q[0] = *(const uint4*)(x0);
        a0.q[1] = *(const uint4*)(x0 + 16);
        const uint16_t* x1 = X + (size_t)(mrow + 16) * D_ + kk + kb;
        a1.q[0] = *(const uint4*)(x1);
        a1.q[1] = *(const uint4*)(x1 + 16);

        #pragma unroll
        for (int nc = 0; nc < 4; ++nc) {
            AFrag bf;
            const int bn = wc * 64 + nc * 16 + (lane & 15);      // lane%16 == N index
            #pragma unroll
            for (int s = 0; s < 16; ++s)
                bf.v[s] = __builtin_bit_cast(__bf16, lb[kb + (s & 7) + ((s >> 3) << 4)][bn]);
            acc[0][nc] = wmma_bf16(a0.v, bf.v, acc[0][nc]);
            acc[1][nc] = wmma_bf16(a1.v, bf.v, acc[1][nc]);
        }
    }

    // Epilogue: C frag element g lives at (M = g + 8*(lane/16), N = lane%16)
    #pragma unroll
    for (int mi = 0; mi < 2; ++mi)
        #pragma unroll
        for (int nc = 0; nc < 4; ++nc)
            #pragma unroll
            for (int g = 0; g < 8; ++g) {
                int row = m_base + mi * 16 + g + 8 * (lane >> 4);
                int col = n_base + nc * 16 + (lane & 15);
                float v = acc[mi][nc][g] + bias[col];
                if (mode == 0) {
                    v = fmaxf(v, 0.f) * scale;
                    int b = row >> 11, s = row & (S_ - 1);
                    int h = col >> 6,  dh = col & 63;
                    out_bf[(((size_t)(b * H_ + h) * S_ + s) << 6) + dh] = f2bf(v);
                } else if (mode == 1) {
                    v = fmaxf(v, 0.f);
                    int b = row >> 11, s = row & (S_ - 1);
                    int h = col >> 6,  dh = col & 63;
                    out_bf[((size_t)(b * H_ + h) * HD_ + dh) * S_ + s] = f2bf(v);
                } else {
                    v += residual[(size_t)row * D_ + col];
                    out_f[(size_t)row * D_ + col] = v;
                }
            }
}

// ---------------------------------------------------------------------------
// Kernel 4: flash attention. Q:(B,H,S,64) bf16 (pre-scaled 1/8), Kt:(B,H,64,S),
// V:(B,H,S,64). Each wave: 16 query rows, online softmax over 32-key chunks.
// Output O:(B,S,U) bf16 for the out-projection GEMM.
// ---------------------------------------------------------------------------
__global__ void __launch_bounds__(256)
k_attn(const uint16_t* __restrict__ Qb, const uint16_t* __restrict__ Ktb,
       const uint16_t* __restrict__ Vb, uint16_t* __restrict__ Ob) {
    __shared__ __align__(16) uint16_t lp[8][16][32];    // per-wave P tile, 8 KB

    const int wid  = threadIdx.x >> 5;
    const int lane = threadIdx.x & 31;
    const int b = blockIdx.z, h = blockIdx.y;
    const int q0 = blockIdx.x * 128 + wid * 16;
    const size_t bh = (size_t)(b * H_ + h);
    const uint16_t* Qp  = Qb  + bh * S_ * HD_;
    const uint16_t* Ktp = Ktb + bh * HD_ * S_;
    const uint16_t* Vp  = Vb  + bh * S_ * HD_;

    const int mrow = q0 + (lane & 15);
    const int kb   = (lane >> 4) * 8;

    AFrag aq0, aq1;                       // Q rows, K-dim 0..31 and 32..63
    {
        const uint16_t* p = Qp + (size_t)mrow * HD_;
        aq0.q[0] = *(const uint4*)(p + kb);
        aq0.q[1] = *(const uint4*)(p + kb + 16);
        aq1.q[0] = *(const uint4*)(p + 32 + kb);
        aq1.q[1] = *(const uint4*)(p + 32 + kb + 16);
    }

    f32x8 acc[4];
    #pragma unroll
    for (int nc = 0; nc < 4; ++nc) acc[nc] = fzero();
    float mrun[8], lrun[8];
    #pragma unroll
    for (int g = 0; g < 8; ++g) { mrun[g] = -3.0e38f; lrun[g] = 0.f; }

    for (int c = 0; c < S_; c += 32) {
        // ---- scores: two 16-key tiles, each 2 WMMAs over head_dim=64 ----
        f32x8 sc[2];
        #pragma unroll
        for (int t = 0; t < 2; ++t) {
            const int key0 = c + t * 16;
            AFrag bk0, bk1;
            const int kn = key0 + (lane & 15);                           // N=key
            #pragma unroll
            for (int s = 0; s < 16; ++s) {
                const int k = kb + (s & 7) + ((s >> 3) << 4);
                bk0.v[s] = __builtin_bit_cast(__bf16, Ktp[(size_t)k * S_ + kn]);
                bk1.v[s] = __builtin_bit_cast(__bf16, Ktp[(size_t)(32 + k) * S_ + kn]);
            }
            f32x8 z = fzero();
            z = wmma_bf16(aq0.v, bk0.v, z);
            z = wmma_bf16(aq1.v, bk1.v, z);
            sc[t] = z;
        }

        // ---- row max across this 32-key chunk (butterfly within 16 lanes) ----
        float cm[8];
        #pragma unroll
        for (int g = 0; g < 8; ++g) cm[g] = fmaxf(sc[0][g], sc[1][g]);
        #pragma unroll
        for (int off = 1; off < 16; off <<= 1)
            #pragma unroll
            for (int g = 0; g < 8; ++g)
                cm[g] = fmaxf(cm[g], __shfl_xor(cm[g], off, 32));

        float mnew[8], alpha[8], rs[8];
        #pragma unroll
        for (int g = 0; g < 8; ++g) {
            mnew[g]  = fmaxf(mrun[g], cm[g]);
            alpha[g] = __expf(mrun[g] - mnew[g]);
            rs[g] = 0.f;
        }

        // ---- p = exp(s - m), stash bf16 P tile in LDS (A-frag staging) ----
        #pragma unroll
        for (int t = 0; t < 2; ++t)
            #pragma unroll
            for (int g = 0; g < 8; ++g) {
                float p = __expf(sc[t][g] - mnew[g]);
                rs[g] += p;
                lp[wid][g + 8 * (lane >> 4)][t * 16 + (lane & 15)] = f2bf(p);
            }

        #pragma unroll
        for (int off = 1; off < 16; off <<= 1)
            #pragma unroll
            for (int g = 0; g < 8; ++g)
                rs[g] += __shfl_xor(rs[g], off, 32);

        #pragma unroll
        for (int g = 0; g < 8; ++g) {
            lrun[g] = lrun[g] * alpha[g] + rs[g];
            mrun[g] = mnew[g];
        }
        #pragma unroll
        for (int nc = 0; nc < 4; ++nc)
            #pragma unroll
            for (int g = 0; g < 8; ++g)
                acc[nc][g] *= alpha[g];

        __builtin_amdgcn_wave_barrier();
        asm volatile("s_wait_dscnt 0" ::: "memory");   // same-wave LDS RAW order

        // ---- read P back in A-fragment layout, P(16x32) @ V(32x64) ----
        AFrag ap;
        {
            const uint16_t* pp = &lp[wid][lane & 15][kb];
            ap.q[0] = *(const uint4*)(pp);
            ap.q[1] = *(const uint4*)(pp + 16);
        }
        #pragma unroll
        for (int nc = 0; nc < 4; ++nc) {
            AFrag bv;
            #pragma unroll
            for (int s = 0; s < 16; ++s)
                bv.v[s] = __builtin_bit_cast(__bf16, Vp[(size_t)(c + kb + (s & 7) + ((s >> 3) << 4)) * HD_ + nc * 16 + (lane & 15)]);
            acc[nc] = wmma_bf16(ap.v, bv.v, acc[nc]);
        }
    }

    // ---- normalize and store O (B,S,U) bf16 ----
    float inv[8];
    #pragma unroll
    for (int g = 0; g < 8; ++g) inv[g] = 1.0f / lrun[g];
    #pragma unroll
    for (int nc = 0; nc < 4; ++nc)
        #pragma unroll
        for (int g = 0; g < 8; ++g) {
            int s   = q0 + g + 8 * (lane >> 4);
            int col = h * HD_ + nc * 16 + (lane & 15);
            Ob[(size_t)(b * S_ + s) * U_ + col] = f2bf(acc[nc][g] * inv[g]);
        }
}

// ---------------------------------------------------------------------------
// Kernel 5: LayerNorm over D + transposed store to (B, D, S)
// ---------------------------------------------------------------------------
__global__ void __launch_bounds__(256)
k_layernorm(const float* __restrict__ z, const float* __restrict__ gamma,
            const float* __restrict__ beta, float* __restrict__ out) {
    int row = blockIdx.x;                 // b*S + s
    int b = row >> 11, s = row & (S_ - 1);
    const float* x = z + (size_t)row * D_;

    float v[4], sum = 0.f, sq = 0.f;
    #pragma unroll
    for (int i = 0; i < 4; ++i) {
        v[i] = x[threadIdx.x + 256 * i];
        sum += v[i];
        sq  += v[i] * v[i];
    }
    #pragma unroll
    for (int off = 16; off > 0; off >>= 1) {
        sum += __shfl_down(sum, off, 32);
        sq  += __shfl_down(sq,  off, 32);
    }
    __shared__ float ls[8], ls2[8];
    int wave = threadIdx.x >> 5, lane = threadIdx.x & 31;
    if (lane == 0) { ls[wave] = sum; ls2[wave] = sq; }
    __syncthreads();
    if (threadIdx.x == 0) {
        float a = 0.f, c2 = 0.f;
        #pragma unroll
        for (int i = 0; i < 8; ++i) { a += ls[i]; c2 += ls2[i]; }
        ls[0] = a; ls2[0] = c2;
    }
    __syncthreads();
    float mean = ls[0] * (1.0f / D_);
    float var  = ls2[0] * (1.0f / D_) - mean * mean;
    float rstd = rsqrtf(var + 1e-5f);
    #pragma unroll
    for (int i = 0; i < 4; ++i) {
        int d = threadIdx.x + 256 * i;
        float y = (v[i] - mean) * rstd * gamma[d] + beta[d];
        out[((size_t)b * D_ + d) * S_ + s] = y;
    }
}

// ---------------------------------------------------------------------------
// Workspace layout (bytes)
// ---------------------------------------------------------------------------
static constexpr size_t OFF_XF = 0;                                  // fp32 x  (16 MB)
static constexpr size_t OFF_XB = OFF_XF + (size_t)NROW * D_ * 4;     // bf16 x  ( 8 MB)
static constexpr size_t OFF_WQ = OFF_XB + (size_t)NROW * D_ * 2;
static constexpr size_t OFF_WK = OFF_WQ + (size_t)D_ * U_ * 2;
static constexpr size_t OFF_WV = OFF_WK + (size_t)D_ * U_ * 2;
static constexpr size_t OFF_WO = OFF_WV + (size_t)D_ * U_ * 2;
static constexpr size_t OFF_Q  = OFF_WO + (size_t)D_ * U_ * 2;       // (B,H,S,hd)
static constexpr size_t OFF_KT = OFF_Q  + (size_t)NROW * U_ * 2;     // (B,H,hd,S)
static constexpr size_t OFF_V  = OFF_KT + (size_t)NROW * U_ * 2;     // (B,H,S,hd)
static constexpr size_t OFF_O  = OFF_V  + (size_t)NROW * U_ * 2;     // (B,S,U)
static constexpr size_t OFF_Z  = OFF_O  + (size_t)NROW * U_ * 2;     // fp32 pre-LN

extern "C" void kernel_launch(void* const* d_in, const int* in_sizes, int n_in,
                              void* d_out, int out_size, void* d_ws, size_t ws_size,
                              hipStream_t stream) {
    (void)in_sizes; (void)n_in; (void)out_size; (void)ws_size;
    const float* inp = (const float*)d_in[0];
    const float* Wq  = (const float*)d_in[1];
    const float* bq  = (const float*)d_in[2];
    const float* Wk  = (const float*)d_in[3];
    const float* bk  = (const float*)d_in[4];
    const float* Wv  = (const float*)d_in[5];
    const float* bv  = (const float*)d_in[6];
    const float* Wo  = (const float*)d_in[7];
    const float* bo  = (const float*)d_in[8];
    const float* gamma = (const float*)d_in[9];
    const float* beta  = (const float*)d_in[10];
    float* out = (float*)d_out;

    char* ws = (char*)d_ws;
    float*    x_f   = (float*)(ws + OFF_XF);
    uint16_t* x_b   = (uint16_t*)(ws + OFF_XB);
    uint16_t* wq_b  = (uint16_t*)(ws + OFF_WQ);
    uint16_t* wk_b  = (uint16_t*)(ws + OFF_WK);
    uint16_t* wv_b  = (uint16_t*)(ws + OFF_WV);
    uint16_t* wo_b  = (uint16_t*)(ws + OFF_WO);
    uint16_t* q_b   = (uint16_t*)(ws + OFF_Q);
    uint16_t* kt_b  = (uint16_t*)(ws + OFF_KT);
    uint16_t* v_b   = (uint16_t*)(ws + OFF_V);
    uint16_t* o_b   = (uint16_t*)(ws + OFF_O);
    float*    z_f   = (float*)(ws + OFF_Z);

    // 1) transpose + bf16 cast of activations
    k_transpose<<<dim3(D_ / 32, S_ / 32, B_), dim3(32, 8), 0, stream>>>(inp, x_f, x_b);

    // 2) weight casts
    const int nW = D_ * U_;
    k_cast<<<(nW + 255) / 256, 256, 0, stream>>>(Wq, wq_b, nW);
    k_cast<<<(nW + 255) / 256, 256, 0, stream>>>(Wk, wk_b, nW);
    k_cast<<<(nW + 255) / 256, 256, 0, stream>>>(Wv, wv_b, nW);
    k_cast<<<(nW + 255) / 256, 256, 0, stream>>>(Wo, wo_b, nW);

    // 3) Q/K/V projections (WMMA bf16). Q pre-scaled by 1/sqrt(hd)=0.125.
    dim3 ggrid(U_ / 256, NROW / 64);
    k_gemm<<<ggrid, 256, 0, stream>>>(x_b, wq_b, bq, nullptr, q_b,  nullptr, 0, 0.125f);
    k_gemm<<<ggrid, 256, 0, stream>>>(x_b, wk_b, bk, nullptr, kt_b, nullptr, 1, 1.0f);
    k_gemm<<<ggrid, 256, 0, stream>>>(x_b, wv_b, bv, nullptr, v_b,  nullptr, 0, 1.0f);

    // 4) flash attention
    k_attn<<<dim3(S_ / 128, H_, B_), 256, 0, stream>>>(q_b, kt_b, v_b, o_b);

    // 5) output projection + bias + residual (fp32 out)
    k_gemm<<<dim3(D_ / 256, NROW / 64), 256, 0, stream>>>(o_b, wo_b, bo, x_f, nullptr, z_f, 2, 1.0f);

    // 6) LayerNorm + transpose back to (B, D, S)
    k_layernorm<<<NROW, 256, 0, stream>>>(z_f, gamma, beta, out);
}
